// Attention_23785528886091
// MI455X (gfx1250) — matrix operand order repaired and checked by reading, not compiled, not run on hardware
//
#include <hip/hip_runtime.h>
#include <hip/hip_bf16.h>

// ---------------------------------------------------------------------------
// Tiny-attention (B=131072, P=3, D=128, H=4, DH=32) for MI455X (gfx1250).
//
// Roofline: ~52 GFLOP vs ~403 MB HBM traffic (17us @ 23.3 TB/s) -> need
// ~3 PFLOP/s to stay memory bound. Only the K=32 bf16 WMMA path gets close;
// fp32 WMMA (K=4) would be ~8x compute bound. So: bf16 operands, f32 accum.
// X tile is DMA'd into LDS by the Tensor Data Mover (one issue per block),
// tracked with TENSORcnt; tail-block rows are zeroed by TDM OOB semantics.
// ---------------------------------------------------------------------------

typedef __bf16 bf16_t;
typedef __attribute__((ext_vector_type(16))) __bf16 v16bf;
typedef __attribute__((ext_vector_type(8)))  __bf16 v8bf;
typedef __attribute__((ext_vector_type(4)))  __bf16 v4bf;
typedef __attribute__((ext_vector_type(8)))  float  v8f;
typedef __attribute__((ext_vector_type(4)))  unsigned int u32x4;
typedef __attribute__((ext_vector_type(8)))  int i32x8;
typedef __attribute__((ext_vector_type(4)))  int i32x4;

#define PP        3
#define DD        128
#define HH        4
#define DHH       32
#define F_QKV     384            // 3 * H * DH fused K|Q|V features
#define ROWS      48             // 16 batches * 3 positions = 3 WMMA row tiles
#define THREADS   128            // 4 wave32s
#define QKV_FRAGS (24 * 4)       // 24 col tiles x 4 k tiles
#define WO_FRAGS  (8 * 4)
#define FRAG_ELEMS 512           // 32 lanes * 16 bf16
#define QKV_WS_ELEMS (QKV_FRAGS * FRAG_ELEMS)   // 49152
#define WO_WS_ELEMS  (WO_FRAGS  * FRAG_ELEMS)   // 16384

#if defined(__has_builtin)
#if __has_builtin(__builtin_amdgcn_tensor_load_to_lds) && \
    __has_builtin(__builtin_amdgcn_s_wait_tensorcnt)
#define HAVE_TDM 1
#endif
#endif

// ---------------------------------------------------------------------------
// Pre-swizzle weights into bf16 WMMA B-fragments (K x 16 tile, per-lane
// contiguous 16 elements). Contiguous k order (not the ISA table order): lanes 0-15 cover
// K = kt*32 + 0..15 (col N = lane&15), lanes 16-31 cover K = kt*32 + 16..31.
// Element e of the per-lane run is K = kt*32 + (lane>>4)*16 + e.
// ---------------------------------------------------------------------------
__global__ void prep_weights_kernel(const float* __restrict__ Wk,
                                    const float* __restrict__ Wq,
                                    const float* __restrict__ Wv,
                                    const float* __restrict__ Wo,
                                    bf16_t* __restrict__ ws)
{
    int tid = blockIdx.x * blockDim.x + threadIdx.x;
    if (tid < QKV_WS_ELEMS) {
        int frag = tid >> 9;              // 512 elems per fragment
        int lane = (tid >> 4) & 31;
        int e    = tid & 15;
        int ct = frag >> 2, kt = frag & 3;
        int f = ct * 16 + (lane & 15);    // fused feature in [0,384)
        int k = kt * 32 + ((lane >> 4) << 4) + e;
        const float* W = (f < 128) ? Wk : (f < 256) ? Wq : Wv;
        // W_{proj}[head, h, d] flattened: (f & 127) * 128 + k
        ws[tid] = (bf16_t)W[(f & 127) * DD + k];
    } else if (tid < QKV_WS_ELEMS + WO_WS_ELEMS) {
        int t2 = tid - QKV_WS_ELEMS;
        int frag = t2 >> 9;
        int lane = (t2 >> 4) & 31;
        int e    = t2 & 15;
        int ct = frag >> 2, kt = frag & 3;
        int d = ct * 16 + (lane & 15);
        int k = kt * 32 + ((lane >> 4) << 4) + e;
        ws[QKV_WS_ELEMS + t2] = (bf16_t)Wo[d * DD + k];   // WoT[k][d]
    }
}

// A-fragment from a row-major bf16 LDS tile (contiguous K order, matches B):
// lane<16  holds K = k0+0..15
// lane>=16 holds K = k0+16..31
__device__ __forceinline__ v16bf load_a_frag(const bf16_t* rowptr, int k0, int hi)
{
    v8bf lo = *(const v8bf*)(rowptr + k0 + hi * 16);
    v8bf up = *(const v8bf*)(rowptr + k0 + hi * 16 + 8);
    v16bf a;
#pragma unroll
    for (int e = 0; e < 8; ++e) { a[e] = lo[e]; a[8 + e] = up[e]; }
    return a;
}

__device__ __forceinline__ v16bf load_b_frag(const bf16_t* base, int frag, int lane)
{
    return *(const v16bf*)(base + ((frag * 32 + lane) << 4));   // 32B aligned
}

__global__ __launch_bounds__(THREADS)
void attn_fused_kernel(const float* __restrict__ x,
                       const bf16_t* __restrict__ wsfrag,
                       float* __restrict__ out,
                       int nrows)
{
    __shared__ alignas(16) bf16_t sX[ROWS][DD];        // 12 KB
    __shared__ alignas(16) bf16_t sQKV[ROWS][F_QKV];   // 36 KB (also f32 X stage)
    __shared__ alignas(16) bf16_t sZ[ROWS][DD];        // 12 KB

    const int tid  = threadIdx.x;
    const int lane = tid & 31;
    const int wave = tid >> 5;
    const int hi   = lane >> 4;            // half-wave selector
    const int m    = lane & 15;            // row within tile (A) / col (C,D)
    const long row0 = (long)blockIdx.x * ROWS;
    const int rows_here = (int)((nrows - row0) < ROWS ? (nrows - row0) : ROWS);

    // ---- Phase 1: stage X tile (48 x 128 f32) and convert to bf16 --------
    float* stage = (float*)&sQKV[0][0];    // 24 KB overlay, dead until phase 2
#if HAVE_TDM
    if (wave == 0) {
        // Tensor DMA descriptor (2D): tile 128 x 48 of 4-byte data, rows_here valid.
        unsigned long long gaddr =
            (unsigned long long)(uintptr_t)x + (unsigned long long)row0 * (DD * 4);
        unsigned lds_off = (unsigned)(uintptr_t)stage;  // low 32b of LDS aperture
        u32x4 g0 = {};
        g0[0] = 1u;                                    // count=1 valid descriptor
        g0[1] = lds_off;                               // lds_addr
        g0[2] = (unsigned)gaddr;                       // global_addr[31:0]
        g0[3] = (unsigned)((gaddr >> 32) & 0x01FFFFFFull) | 0x80000000u; // [56:32] | type=2
        i32x8 g1 = {};
        g1[0] = (int)(2u << 16);                       // data_size = 2 -> 4 bytes
        g1[1] = (int)((unsigned)DD << 16);             // tensor_dim0 = 128
        g1[2] = (int)((unsigned)rows_here << 16);      // tensor_dim1 = valid rows
        g1[3] = (int)((unsigned)DD << 16);             // tile_dim0 = 128
        g1[4] = (int)(unsigned)ROWS;                   // tile_dim1 = 48
        g1[5] = (int)(unsigned)DD;                     // tensor_dim0_stride = 128
        i32x4 z4 = {};
#if __has_include(<hip/amd_detail/amd_gfx1250_TDM.h>)
        i32x8 z8 = {};
        __builtin_amdgcn_tensor_load_to_lds(g0, g1, z4, z4, z8, 0);  // clang-23 form
#else
        __builtin_amdgcn_tensor_load_to_lds(g0, g1, z4, z4, 0);      // ROCm 7.2 form
#endif
        __builtin_amdgcn_s_wait_tensorcnt(0);
    }
    __syncthreads();                       // TDM data visible to all waves
    {
        const float4* sv = (const float4*)stage;
#pragma unroll
        for (int it = 0; it < (ROWS * DD / 4) / THREADS; ++it) {   // 12 iters
            int idx = it * THREADS + tid;
            float4 v = sv[idx];
            int r  = idx >> 5;             // 32 float4 per row
            int c4 = idx & 31;
            v4bf cv = { (bf16_t)v.x, (bf16_t)v.y, (bf16_t)v.z, (bf16_t)v.w };
            *(v4bf*)&sX[r][c4 * 4] = cv;
        }
    }
#else
    {
        const float4* xv = (const float4*)x;
#pragma unroll
        for (int it = 0; it < (ROWS * DD / 4) / THREADS; ++it) {   // 12 iters
            int idx = it * THREADS + tid;
            int r   = idx >> 5;
            int c4  = idx & 31;
            float4 v = {0.f, 0.f, 0.f, 0.f};
            if (r < rows_here) v = xv[(row0 + r) * (DD / 4) + c4];
            v4bf cv = { (bf16_t)v.x, (bf16_t)v.y, (bf16_t)v.z, (bf16_t)v.w };
            *(v4bf*)&sX[r][c4 * 4] = cv;
        }
    }
#endif
    __syncthreads();

    // ---- Phase 2: GEMM1  QKV[48,384] = X[48,128] x WqkvT[128,384] --------
    // A-fragments hoisted per row stripe (reused across 6 column tiles).
    for (int rt = 0; rt < 3; ++rt) {
        const bf16_t* arow = &sX[rt * 16 + m][0];
        v16bf a0 = load_a_frag(arow,  0, hi);
        v16bf a1 = load_a_frag(arow, 32, hi);
        v16bf a2 = load_a_frag(arow, 64, hi);
        v16bf a3 = load_a_frag(arow, 96, hi);
        for (int ct = wave; ct < 24; ct += 4) {          // wave-uniform trip=6
            v8f acc = {};
            acc = __builtin_amdgcn_wmma_f32_16x16x32_bf16(
                      false, a0, false, load_b_frag(wsfrag, ct * 4 + 0, lane),
                      (short)0, acc, false, false);
            acc = __builtin_amdgcn_wmma_f32_16x16x32_bf16(
                      false, a1, false, load_b_frag(wsfrag, ct * 4 + 1, lane),
                      (short)0, acc, false, false);
            acc = __builtin_amdgcn_wmma_f32_16x16x32_bf16(
                      false, a2, false, load_b_frag(wsfrag, ct * 4 + 2, lane),
                      (short)0, acc, false, false);
            acc = __builtin_amdgcn_wmma_f32_16x16x32_bf16(
                      false, a3, false, load_b_frag(wsfrag, ct * 4 + 3, lane),
                      (short)0, acc, false, false);
#pragma unroll
            for (int r = 0; r < 8; ++r)                  // D tile -> LDS bf16
                sQKV[rt * 16 + r + hi * 8][ct * 16 + m] = (bf16_t)acc[r];
        }
    }
    __syncthreads();

    // ---- Phase 3: per-(batch,head) causal 3x3 attention ------------------
    if (tid < 16 * HH) {
        int g    = tid >> 2;                 // batch within block
        int head = tid & 3;
        int rb   = g * 3;
        if (rb + 2 < rows_here) {
            const int ok = head * DHH;               // K features
            const int oq = 128 + head * DHH;         // Q features
            const int ov = 256 + head * DHH;         // V features
            float s00 = 0.f, s10 = 0.f, s11 = 0.f, s20 = 0.f, s21 = 0.f, s22 = 0.f;
#pragma unroll
            for (int hc = 0; hc < 4; ++hc) {
                v8bf k0 = *(const v8bf*)&sQKV[rb + 0][ok + hc * 8];
                v8bf k1 = *(const v8bf*)&sQKV[rb + 1][ok + hc * 8];
                v8bf k2 = *(const v8bf*)&sQKV[rb + 2][ok + hc * 8];
                v8bf q0 = *(const v8bf*)&sQKV[rb + 0][oq + hc * 8];
                v8bf q1 = *(const v8bf*)&sQKV[rb + 1][oq + hc * 8];
                v8bf q2 = *(const v8bf*)&sQKV[rb + 2][oq + hc * 8];
#pragma unroll
                for (int e = 0; e < 8; ++e) {
                    float fk0 = (float)k0[e], fk1 = (float)k1[e], fk2 = (float)k2[e];
                    float fq0 = (float)q0[e], fq1 = (float)q1[e], fq2 = (float)q2[e];
                    s00 += fk0 * fq0;
                    s10 += fk0 * fq1; s11 += fk1 * fq1;
                    s20 += fk0 * fq2; s21 += fk1 * fq2; s22 += fk2 * fq2;
                }
            }
            const float scale = 0.17677669529663687f;   // 1/sqrt(32)
            (void)s00;   // q=0: softmax over single element -> attn row0 = [1,0,0]
            float a10, a11, a20, a21, a22;
            {
                float u0 = s10 * scale, u1 = s11 * scale;
                float mx = fmaxf(u0, u1);
                float e0 = __expf(u0 - mx), e1 = __expf(u1 - mx);
                float inv = 1.f / (e0 + e1);
                a10 = e0 * inv; a11 = e1 * inv;
            }
            {
                float u0 = s20 * scale, u1 = s21 * scale, u2 = s22 * scale;
                float mx = fmaxf(fmaxf(u0, u1), u2);
                float e0 = __expf(u0 - mx), e1 = __expf(u1 - mx), e2 = __expf(u2 - mx);
                float inv = 1.f / (e0 + e1 + e2);
                a20 = e0 * inv; a21 = e1 * inv; a22 = e2 * inv;
            }
#pragma unroll
            for (int hc = 0; hc < 4; ++hc) {
                v8bf v0 = *(const v8bf*)&sQKV[rb + 0][ov + hc * 8];
                v8bf v1 = *(const v8bf*)&sQKV[rb + 1][ov + hc * 8];
                v8bf v2 = *(const v8bf*)&sQKV[rb + 2][ov + hc * 8];
                v8bf z0, z1, z2;
#pragma unroll
                for (int e = 0; e < 8; ++e) {
                    float fv0 = (float)v0[e], fv1 = (float)v1[e], fv2 = (float)v2[e];
                    z0[e] = (bf16_t)fv0;
                    z1[e] = (bf16_t)(a10 * fv0 + a11 * fv1);
                    z2[e] = (bf16_t)(a20 * fv0 + a21 * fv1 + a22 * fv2);
                }
                *(v8bf*)&sZ[rb + 0][head * DHH + hc * 8] = z0;
                *(v8bf*)&sZ[rb + 1][head * DHH + hc * 8] = z1;
                *(v8bf*)&sZ[rb + 2][head * DHH + hc * 8] = z2;
            }
        }
    }
    __syncthreads();

    // ---- Phase 4: GEMM2  OUT[48,128] = Z[48,128] x WoT[128,128] ----------
    const bf16_t* wo = wsfrag + QKV_WS_ELEMS;
    for (int rt = 0; rt < 3; ++rt) {
        const bf16_t* arow = &sZ[rt * 16 + m][0];
        v16bf a0 = load_a_frag(arow,  0, hi);
        v16bf a1 = load_a_frag(arow, 32, hi);
        v16bf a2 = load_a_frag(arow, 64, hi);
        v16bf a3 = load_a_frag(arow, 96, hi);
        for (int ct = wave; ct < 8; ct += 4) {           // wave-uniform trip=2
            v8f acc = {};
            acc = __builtin_amdgcn_wmma_f32_16x16x32_bf16(
                      false, a0, false, load_b_frag(wo, ct * 4 + 0, lane),
                      (short)0, acc, false, false);
            acc = __builtin_amdgcn_wmma_f32_16x16x32_bf16(
                      false, a1, false, load_b_frag(wo, ct * 4 + 1, lane),
                      (short)0, acc, false, false);
            acc = __builtin_amdgcn_wmma_f32_16x16x32_bf16(
                      false, a2, false, load_b_frag(wo, ct * 4 + 2, lane),
                      (short)0, acc, false, false);
            acc = __builtin_amdgcn_wmma_f32_16x16x32_bf16(
                      false, a3, false, load_b_frag(wo, ct * 4 + 3, lane),
                      (short)0, acc, false, false);
#pragma unroll
            for (int r = 0; r < 8; ++r) {
                long rr = row0 + rt * 16 + r + hi * 8;
                if (rr < nrows)
                    out[rr * DD + ct * 16 + m] = acc[r];
            }
        }
    }
}

extern "C" void kernel_launch(void* const* d_in, const int* in_sizes, int n_in,
                              void* d_out, int out_size, void* d_ws, size_t ws_size,
                              hipStream_t stream)
{
    const float* x  = (const float*)d_in[0];
    const float* Wk = (const float*)d_in[1];
    const float* Wq = (const float*)d_in[2];
    const float* Wv = (const float*)d_in[3];
    const float* Wo = (const float*)d_in[4];
    float*  out = (float*)d_out;
    bf16_t* ws  = (bf16_t*)d_ws;

    const int nrows = in_sizes[0] / DD;              // B * P

    const int prep_elems = QKV_WS_ELEMS + WO_WS_ELEMS;  // 65536
    prep_weights_kernel<<<(prep_elems + 255) / 256, 256, 0, stream>>>(Wk, Wq, Wv, Wo, ws);

    const int nblocks = (nrows + ROWS - 1) / ROWS;
    attn_fused_kernel<<<nblocks, THREADS, 0, stream>>>(x, ws, out, nrows);
}
